// GCN_86294482911287
// MI455X (gfx1250) — matrix operand order repaired and checked by reading, not compiled, not run on hardware
//
#include <hip/hip_runtime.h>
#include <hip/hip_bf16.h>

typedef __attribute__((ext_vector_type(16))) _Float16 v16h;
typedef __attribute__((ext_vector_type(8)))  float    v8f;

#define N_LAYERS 7
#define X_FEATS  5
#define EPS_GEN  1e-7f
#define BN_EPS   1e-5f
#define DROP_KEEP_SCALE 1.25f            // 1/(1-0.2)
#define DROP_THRESH 858993459u           // 0.2 * 2^32

static inline int ceil16(int v) { return (v + 15) & ~15; }
static inline int ceil32(int v) { return (v + 31) & ~31; }

// ---------------------------------------------------------------------------
// helpers
// ---------------------------------------------------------------------------
__device__ __forceinline__ uint32_t rotl32(uint32_t x, int r) {
  return (x << r) | (x >> (32 - r));
}

// threefry2x32-20 (deterministic per-layer dropout PRNG)
__device__ __forceinline__ uint32_t threefry2x32(uint32_t k0, uint32_t k1,
                                                 uint32_t c0, uint32_t c1) {
  uint32_t ks[3] = {k0, k1, 0x1BD11BDAu ^ k0 ^ k1};
  uint32_t x0 = c0 + ks[0];
  uint32_t x1 = c1 + ks[1];
  const int rotA[4] = {13, 15, 26, 6};
  const int rotB[4] = {17, 29, 16, 24};
#pragma unroll
  for (int blk = 0; blk < 5; ++blk) {
    const int* rot = (blk & 1) ? rotB : rotA;
#pragma unroll
    for (int r = 0; r < 4; ++r) {
      x0 += x1;
      x1 = rotl32(x1, rot[r]);
      x1 ^= x0;
    }
    int j = blk + 1;
    x0 += ks[j % 3];
    x1 += ks[(j + 1) % 3] + (uint32_t)j;
  }
  return x0;
}

// monotone float<->uint encoding so atomicMax(u32) == float max
__device__ __forceinline__ uint32_t enc_f32(float f) {
  uint32_t u = __float_as_uint(f);
  return (u & 0x80000000u) ? ~u : (u | 0x80000000u);
}
__device__ __forceinline__ float dec_f32(uint32_t k) {
  uint32_t u = (k & 0x80000000u) ? (k & 0x7FFFFFFFu) : ~k;
  return __uint_as_float(u);
}
#define ENC_NEG_INF 0x007FFFFFu  // enc(-inf)

// ---------------------------------------------------------------------------
// WMMA tile loads (layouts per CDNA5 ISA 7.12.2, wave32).
// All loads unconditional: buffers are stride-padded and weights zero-padded,
// so no EXEC juggling and everything vectorizes to b128 loads.
// ---------------------------------------------------------------------------
// A tile 16x32 f16: lane L -> M=L&15, kbase=(L>=16)?8:0.
// half j -> K = k0 + (j>>3)*16 + kbase + (j&7)  => two contiguous 8-float runs.
template <int MODEA>
__device__ __forceinline__ v16h load_a_tile(const float* __restrict__ A, int sA,
                                            int mtile, int k0, int lane,
                                            const float* __restrict__ asc,
                                            const float* __restrict__ ash) {
  int m = mtile * 16 + (lane & 15);
  int kb = (lane >= 16) ? 8 : 0;
  const float* p = A + (size_t)m * (size_t)sA + k0 + kb;
  float4 q0 = *(const float4*)(p + 0);
  float4 q1 = *(const float4*)(p + 4);
  float4 q2 = *(const float4*)(p + 16);
  float4 q3 = *(const float4*)(p + 20);
  float f[16] = {q0.x, q0.y, q0.z, q0.w, q1.x, q1.y, q1.z, q1.w,
                 q2.x, q2.y, q2.z, q2.w, q3.x, q3.y, q3.z, q3.w};
  v16h a;
#pragma unroll
  for (int j = 0; j < 16; ++j) {
    float v = f[j];
    if (MODEA == 1) {  // fused BatchNorm (per-K-column affine) + ReLU
      int k = k0 + ((j >> 3) * 16) + kb + (j & 7);
      v = fmaxf(v * asc[k] + ash[k], 0.0f);
    }
    a[j] = (_Float16)v;
  }
  return a;
}

// B tile 32x16 f16 from TRANSPOSED padded weights WT[n][k] (n-major, KP cols):
// lane L -> N=L&15, kbase=(L>=16)?8:0; half j -> K = k0+(j>>3)*16+kbase+(j&7).
__device__ __forceinline__ v16h load_b_tile(const float* __restrict__ WT,
                                            int KP, int ntile, int k0,
                                            int lane) {
  int n = ntile * 16 + (lane & 15);
  int kb = (lane >= 16) ? 8 : 0;
  const float* p = WT + (size_t)n * (size_t)KP + k0 + kb;
  float4 q0 = *(const float4*)(p + 0);
  float4 q1 = *(const float4*)(p + 4);
  float4 q2 = *(const float4*)(p + 16);
  float4 q3 = *(const float4*)(p + 20);
  float f[16] = {q0.x, q0.y, q0.z, q0.w, q1.x, q1.y, q1.z, q1.w,
                 q2.x, q2.y, q2.z, q2.w, q3.x, q3.y, q3.z, q3.w};
  v16h b;
#pragma unroll
  for (int j = 0; j < 16; ++j) b[j] = (_Float16)f[j];
  return b;
}

// ---------------------------------------------------------------------------
// generic WMMA GEMM: C[N x *] = epi( modeA(A) @ WT^T ), N % 16 == 0
// KP = ceil32(K) = A row stride. sC = C row stride. grid.y*16 <= sC.
// MODEA: 0 plain, 1 BN+ReLU on A columns.  EPI: 0 plain, 1 +column stats.
// ---------------------------------------------------------------------------
template <int MODEA, int EPI>
__global__ void gemm_wmma_kernel(const float* __restrict__ A,
                                 const float* __restrict__ WT,
                                 float* __restrict__ C, int KP, int sC,
                                 const float* __restrict__ asc,
                                 const float* __restrict__ ash,
                                 float* __restrict__ colsum,
                                 float* __restrict__ colsq) {
  int lane = threadIdx.x & 31;
  int wave = threadIdx.x >> 5;
  int mtile = blockIdx.x * (blockDim.x >> 5) + wave;
  int ntile = blockIdx.y;

  v8f acc = {};
  for (int k0 = 0; k0 < KP; k0 += 32) {
    v16h a = load_a_tile<MODEA>(A, KP, mtile, k0, lane, asc, ash);
    v16h b = load_b_tile(WT, KP, ntile, k0, lane);
    acc = __builtin_amdgcn_wmma_f32_16x16x32_f16(false, a, false, b, (short)0,
                                                 acc, false, false);
  }

  int n = ntile * 16 + (lane & 15);
  if (EPI == 1) {  // BN statistics: each lane covers 8 rows of column n
    float s = 0.0f, q = 0.0f;
#pragma unroll
    for (int r = 0; r < 8; ++r) {
      s += acc[r];
      q += acc[r] * acc[r];
    }
    atomicAdd(&colsum[n], s);
    atomicAdd(&colsq[n], q);
  }
#pragma unroll
  for (int r = 0; r < 8; ++r) {
    int m = mtile * 16 + ((lane < 16) ? r : (r + 8));
    C[(size_t)m * (size_t)sC + n] = acc[r];
  }
}

// dual GEMM (GENConv node transform): S = relu(A@Wsrc)+eps ; D = A@Wdst
__global__ void gemm_dual_wmma_kernel(const float* __restrict__ A,
                                      const float* __restrict__ WsT,
                                      const float* __restrict__ WdT,
                                      float* __restrict__ S,
                                      float* __restrict__ D, int KP, int sC) {
  int lane = threadIdx.x & 31;
  int wave = threadIdx.x >> 5;
  int mtile = blockIdx.x * (blockDim.x >> 5) + wave;
  int ntile = blockIdx.y;

  v8f accS = {};
  v8f accD = {};
  for (int k0 = 0; k0 < KP; k0 += 32) {
    v16h a = load_a_tile<0>(A, KP, mtile, k0, lane, nullptr, nullptr);
    v16h bs = load_b_tile(WsT, KP, ntile, k0, lane);
    v16h bd = load_b_tile(WdT, KP, ntile, k0, lane);
    accS = __builtin_amdgcn_wmma_f32_16x16x32_f16(false, a, false, bs, (short)0,
                                                  accS, false, false);
    accD = __builtin_amdgcn_wmma_f32_16x16x32_f16(false, a, false, bd, (short)0,
                                                  accD, false, false);
  }
  int n = ntile * 16 + (lane & 15);
#pragma unroll
  for (int r = 0; r < 8; ++r) {
    int m = mtile * 16 + ((lane < 16) ? r : (r + 8));
    size_t idx = (size_t)m * (size_t)sC + n;
    S[idx] = fmaxf(accS[r], 0.0f) + EPS_GEN;
    D[idx] = accD[r];
  }
}

// ---------------------------------------------------------------------------
// prep / edge / elementwise kernels
// ---------------------------------------------------------------------------
__global__ void fill_u32_kernel(uint32_t* __restrict__ p, uint32_t v,
                                size_t n) {
  size_t i = (size_t)blockIdx.x * blockDim.x + threadIdx.x;
  size_t stride = (size_t)gridDim.x * blockDim.x;
  for (; i < n; i += stride) p[i] = v;
}

// WT[n*KP + k] = (k<K && n<CO) ? W[k*CO + n] : 0   (transpose + zero-pad)
__global__ void prep_weightT_kernel(const float* __restrict__ W,
                                    float* __restrict__ WT, int K, int CO,
                                    int KP, int NR) {
  int i = blockIdx.x * blockDim.x + threadIdx.x;
  if (i >= NR * KP) return;
  int n = i / KP, k = i % KP;
  WT[i] = (k < K && n < CO) ? W[(size_t)k * CO + n] : 0.0f;
}

// pack x (N x 5) into stride-32 buffer
__global__ void pack_x_kernel(const float* __restrict__ x,
                              float* __restrict__ XP, int N) {
  int i = blockIdx.x * blockDim.x + threadIdx.x;
  if (i >= N * X_FEATS) return;
  int n = i / X_FEATS, c = i % X_FEATS;
  XP[(size_t)n * 32 + c] = x[i];
}

// pass 1: segment-max of z = t*m (encoded atomicMax), vectorized gather
__global__ void edge_max_kernel(const int* __restrict__ src,
                                const int* __restrict__ dst,
                                const float* __restrict__ S,
                                uint32_t* __restrict__ ZE,
                                const float* __restrict__ tptr, int E, int co,
                                int sp) {
  int e = blockIdx.x * blockDim.x + threadIdx.x;
  if (e >= E) return;
  float t = tptr[0];
  int s = src[e], d = dst[e];
  const float* srow = S + (size_t)s * (size_t)sp;
  uint32_t* drow = ZE + (size_t)d * (size_t)sp;
  for (int c = 0; c < co; c += 4) {
    float4 m4 = *(const float4*)(srow + c);
    atomicMax(&drow[c + 0], enc_f32(t * m4.x));
    atomicMax(&drow[c + 1], enc_f32(t * m4.y));
    atomicMax(&drow[c + 2], enc_f32(t * m4.z));
    atomicMax(&drow[c + 3], enc_f32(t * m4.w));
  }
}

// pass 2: e = exp(z - zmax[dst]); denom += e; num += m*e
__global__ void edge_sum_kernel(const int* __restrict__ src,
                                const int* __restrict__ dst,
                                const float* __restrict__ S,
                                const uint32_t* __restrict__ ZE,
                                float* __restrict__ DEN,
                                float* __restrict__ NUM,
                                const float* __restrict__ tptr, int E, int co,
                                int sp) {
  int e = blockIdx.x * blockDim.x + threadIdx.x;
  if (e >= E) return;
  float t = tptr[0];
  int s = src[e], d = dst[e];
  const float* srow = S + (size_t)s * (size_t)sp;
  size_t dbase = (size_t)d * (size_t)sp;
  for (int c = 0; c < co; c += 4) {
    float4 m4 = *(const float4*)(srow + c);
    float mm[4] = {m4.x, m4.y, m4.z, m4.w};
#pragma unroll
    for (int u = 0; u < 4; ++u) {
      float zm = dec_f32(ZE[dbase + c + u]);
      if (!(fabsf(zm) <= 3.4e38f)) zm = 0.0f;  // isfinite guard
      float ex = __expf(t * mm[u] - zm);
      atomicAdd(&DEN[dbase + c + u], ex);
      atomicAdd(&NUM[dbase + c + u], mm[u] * ex);
    }
  }
}

// pass 3: out = num / max(denom, 1e-16) + (x @ W_dst)   (sp -> ap restride)
__global__ void node_aggr_kernel(const float* __restrict__ NUM,
                                 const float* __restrict__ DEN,
                                 const float* __restrict__ D,
                                 float* __restrict__ OUT, int N, int co, int sp,
                                 int ap) {
  size_t i = (size_t)blockIdx.x * blockDim.x + threadIdx.x;
  if (i >= (size_t)N * co) return;
  int n = (int)(i / co), c = (int)(i % co);
  size_t si = (size_t)n * sp + c;
  OUT[(size_t)n * ap + c] = NUM[si] / fmaxf(DEN[si], 1e-16f) + D[si];
}

// BN finalize: writes ALL 128 entries (zeros beyond C2 -> padded K cols safe)
__global__ void bn_finalize_kernel(const float* __restrict__ colsum,
                                   const float* __restrict__ colsq,
                                   const float* __restrict__ gamma,
                                   const float* __restrict__ beta,
                                   float* __restrict__ scale,
                                   float* __restrict__ shift, int C2,
                                   float invN) {
  int c = blockIdx.x * blockDim.x + threadIdx.x;
  if (c >= 128) return;
  float scl = 0.0f, shf = 0.0f;
  if (c < C2) {
    float mu = colsum[c] * invN;
    float var = colsq[c] * invN - mu * mu;
    float rs = rsqrtf(fmaxf(var, 0.0f) + BN_EPS);
    scl = rs * gamma[c];
    shf = beta[c] - mu * scl;
  }
  scale[c] = scl;
  shift[c] = shf;
}

// concat [H2 | x] + dropout(threefry); H2 stride sp -> out stride so
__global__ void concat_dropout_kernel(const float* __restrict__ H2,
                                      const float* __restrict__ X,
                                      float* __restrict__ OUT, int N, int co,
                                      int cin, int sp, int so, int layer) {
  size_t i = (size_t)blockIdx.x * blockDim.x + threadIdx.x;
  if (i >= (size_t)N * cin) return;
  int n = (int)(i / cin), c = (int)(i % cin);
  float v = (c < co) ? H2[(size_t)n * sp + c]
                     : X[(size_t)n * X_FEATS + (c - co)];
  uint32_t r = threefry2x32(42u, 0x9E3779B9u * (uint32_t)(layer + 1),
                            (uint32_t)i, (uint32_t)layer);
  OUT[(size_t)n * so + c] = (r >= DROP_THRESH) ? v * DROP_KEEP_SCALE : 0.0f;
}

// SAGE: per-edge mean accumulation (13 feats, H stride sh)
__global__ void sage_edge_kernel(const int* __restrict__ src,
                                 const int* __restrict__ dst,
                                 const float* __restrict__ H,
                                 float* __restrict__ cnt,
                                 float* __restrict__ acc, int E, int sh) {
  int e = blockIdx.x * blockDim.x + threadIdx.x;
  if (e >= E) return;
  int s = src[e], d = dst[e];
  atomicAdd(&cnt[d], 1.0f);
  const float* srow = H + (size_t)s * sh;
  float* drow = acc + (size_t)d * 13;
#pragma unroll
  for (int c = 0; c < 13; ++c) atomicAdd(&drow[c], srow[c]);
}

__global__ void sage_node_kernel(const float* __restrict__ H,
                                 const float* __restrict__ cnt,
                                 const float* __restrict__ acc,
                                 const float* __restrict__ Wl,
                                 const float* __restrict__ Wr,
                                 const float* __restrict__ bl,
                                 float* __restrict__ out, int N, int sh) {
  int n = blockIdx.x * blockDim.x + threadIdx.x;
  if (n >= N) return;
  float cc = fmaxf(cnt[n], 1.0f);
  float sum = bl[0];
  const float* arow = acc + (size_t)n * 13;
  const float* hrow = H + (size_t)n * sh;
#pragma unroll
  for (int c = 0; c < 13; ++c) {
    sum += (arow[c] / cc) * Wl[c] + hrow[c] * Wr[c];
  }
  out[n] = sum;
}

// ---------------------------------------------------------------------------
// host orchestration
// ---------------------------------------------------------------------------
extern "C" void kernel_launch(void* const* d_in, const int* in_sizes, int n_in,
                              void* d_out, int out_size, void* d_ws,
                              size_t ws_size, hipStream_t stream) {
  (void)n_in; (void)out_size; (void)ws_size;

  static const int CI[N_LAYERS] = {5, 13, 21, 37, 69, 37, 21};
  static const int CO[N_LAYERS] = {8, 16, 32, 64, 32, 16, 8};

  const float* x = (const float*)d_in[0];
  const int* ei = (const int*)d_in[1];
  const int N = in_sizes[0] / X_FEATS;  // 100000, multiple of 16
  const int E = in_sizes[1] / 2;
  const int* src = ei;
  const int* dst = ei + E;

  // params per jax pytree order: gen[i]: W1,W2,W_dst,W_src,beta,gamma,t ;
  // sage: W_l, W_r, b_l
  const float *W1[N_LAYERS], *W2[N_LAYERS], *Wdst[N_LAYERS], *Wsrc[N_LAYERS];
  const float *beta[N_LAYERS], *gamma[N_LAYERS], *tptr[N_LAYERS];
  int pi = 2;
  for (int i = 0; i < N_LAYERS; ++i) {
    W1[i]    = (const float*)d_in[pi++];
    W2[i]    = (const float*)d_in[pi++];
    Wdst[i]  = (const float*)d_in[pi++];
    Wsrc[i]  = (const float*)d_in[pi++];
    beta[i]  = (const float*)d_in[pi++];
    gamma[i] = (const float*)d_in[pi++];
    tptr[i]  = (const float*)d_in[pi++];
  }
  const float* Wl = (const float*)d_in[pi++];
  const float* Wr = (const float*)d_in[pi++];
  const float* bl = (const float*)d_in[pi++];

  // workspace (floats); every buffer gets tail slack for K-overread
  float* wsf = (float*)d_ws;
  size_t off = 0;
  auto alloc = [&](size_t nf) {
    float* p = wsf + off;
    off += ((nf + 128) + 63) & ~(size_t)63;
    return p;
  };
  float* XP  = alloc((size_t)N * 32);   // x packed, stride 32
  float* HB0 = alloc((size_t)N * 96);   // h buffers, stride = ceil32(ci)
  float* HB1 = alloc((size_t)N * 96);
  float* S   = alloc((size_t)N * 64);   // stride sp = ceil16(co)
  float* D   = alloc((size_t)N * 64);   // also H2
  float* ZE  = alloc((size_t)N * 64);   // encoded zmax; also OUT (stride ap)
  float* DEN = alloc((size_t)N * 64);   // also SAGE cnt
  float* NUM = alloc((size_t)N * 64);   // also SAGE acc
  float* H1  = alloc((size_t)N * 128);  // stride h1p = ceil32(2co)
  float* WsT = alloc(16384);
  float* WdT = alloc(16384);
  float* W1T = alloc(16384);
  float* W2T = alloc(16384);
  float* colsum  = alloc(128);
  float* colsq   = alloc(128);
  float* bnscale = alloc(128);
  float* bnshift = alloc(128);

  const dim3 blk(128);  // 4 waves/block
  const int gx = (N / 16 + 3) / 4;
  const int eb = (E + 255) / 256;

  pack_x_kernel<<<(N * X_FEATS + 255) / 256, 256, 0, stream>>>(x, XP, N);

  const float* hin = XP;
  for (int i = 0; i < N_LAYERS; ++i) {
    const int ci = CI[i], co = CO[i], c2 = 2 * co;
    const int kpi = ceil32(ci);      // A stride / K-pad of input GEMMs
    const int sp  = ceil16(co);      // edge-buffer stride
    const int ap  = ceil32(co);      // OUT stride / K-pad of gemm1
    const int h1p = ceil32(c2);      // H1 stride / K-pad of gemm2
    const size_t nsp = (size_t)N * sp;

    // pack transposed zero-padded weights
    auto prep = [&](const float* W, float* WT, int K, int Cc, int KP, int NR) {
      int tot = NR * KP;
      prep_weightT_kernel<<<(tot + 255) / 256, 256, 0, stream>>>(W, WT, K, Cc,
                                                                 KP, NR);
    };
    prep(Wsrc[i], WsT, ci, co, kpi, sp);
    prep(Wdst[i], WdT, ci, co, kpi, sp);
    prep(W1[i],   W1T, co, c2, ap,  c2);
    prep(W2[i],   W2T, c2, co, h1p, sp);

    // init accumulators
    fill_u32_kernel<<<4096, 256, 0, stream>>>((uint32_t*)ZE, ENC_NEG_INF, nsp);
    fill_u32_kernel<<<4096, 256, 0, stream>>>((uint32_t*)DEN, 0u, nsp);
    fill_u32_kernel<<<4096, 256, 0, stream>>>((uint32_t*)NUM, 0u, nsp);
    fill_u32_kernel<<<1, 256, 0, stream>>>((uint32_t*)colsum, 0u, 128);
    fill_u32_kernel<<<1, 256, 0, stream>>>((uint32_t*)colsq, 0u, 128);

    // node transform: S = relu(h@Wsrc)+eps ; D = h@Wdst  (stride sp)
    dim3 g1(gx, sp / 16);
    gemm_dual_wmma_kernel<<<g1, blk, 0, stream>>>(hin, WsT, WdT, S, D, kpi, sp);

    // scatter softmax over edges
    edge_max_kernel<<<eb, 256, 0, stream>>>(src, dst, S, (uint32_t*)ZE,
                                            tptr[i], E, co, sp);
    edge_sum_kernel<<<eb, 256, 0, stream>>>(src, dst, S, (const uint32_t*)ZE,
                                            DEN, NUM, tptr[i], E, co, sp);
    float* OUT = ZE;  // reuse (zmax dead)
    node_aggr_kernel<<<(int)(((size_t)N * co + 255) / 256), 256, 0, stream>>>(
        NUM, DEN, D, OUT, N, co, sp, ap);

    // MLP: H1 = OUT @ W1 (+ BN column stats)
    dim3 g2(gx, c2 / 16);
    gemm_wmma_kernel<0, 1><<<g2, blk, 0, stream>>>(OUT, W1T, H1, ap, h1p,
                                                   nullptr, nullptr, colsum,
                                                   colsq);
    bn_finalize_kernel<<<1, 128, 0, stream>>>(colsum, colsq, gamma[i], beta[i],
                                              bnscale, bnshift, c2,
                                              1.0f / (float)N);
    // H2 = relu(BN(H1)) @ W2  (BN+ReLU fused into A load) -> D (stride sp)
    dim3 g3(gx, sp / 16);
    gemm_wmma_kernel<1, 0><<<g3, blk, 0, stream>>>(H1, W2T, D, h1p, sp, bnscale,
                                                   bnshift, nullptr, nullptr);

    // concat with x + dropout -> next layer input (stride ceil32(co+5))
    float* hout = (i % 2 == 0) ? HB0 : HB1;
    const int cin = co + X_FEATS;
    const int so = ceil32(cin);
    size_t tot = (size_t)N * cin;
    concat_dropout_kernel<<<(int)((tot + 255) / 256), 256, 0, stream>>>(
        D, x, hout, N, co, cin, sp, so, i);
    hin = hout;
  }

  // SAGEConv(13 -> 1): mean aggregation + root weight (h stride 32)
  float* cnt = DEN;
  float* acc = NUM;
  fill_u32_kernel<<<4096, 256, 0, stream>>>((uint32_t*)cnt, 0u, (size_t)N);
  fill_u32_kernel<<<4096, 256, 0, stream>>>((uint32_t*)acc, 0u, (size_t)N * 13);
  sage_edge_kernel<<<eb, 256, 0, stream>>>(src, dst, hin, cnt, acc, E, 32);
  sage_node_kernel<<<(N + 255) / 256, 256, 0, stream>>>(hin, cnt, acc, Wl, Wr,
                                                        bl, (float*)d_out, N,
                                                        32);
}
